// AttentionDecoder_8589935292
// MI455X (gfx1250) — compile-verified, matrix operand order checked by reading, not run on hardware
//
#include <hip/hip_runtime.h>

// ---------------------------------------------------------------------------
// Types for CDNA5 WMMA (gfx1250, wave32)
// ---------------------------------------------------------------------------
typedef __bf16 v16bf __attribute__((ext_vector_type(16)));
typedef __bf16 v8bf  __attribute__((ext_vector_type(8)));
typedef float  v8f   __attribute__((ext_vector_type(8)));
typedef unsigned short u16t;

static constexpr int Bc = 128;     // batch
static constexpr int Sc = 49;      // spatial
static constexpr int Tc = 16;      // timesteps
static constexpr int Hc = 512;     // hidden
static constexpr int Vc = 32000;   // vocab
static constexpr int Lc = 4;       // gru layers

__device__ __forceinline__ u16t f2bf(float f) {
    unsigned int u = __float_as_uint(f);
    u += 0x7fffu + ((u >> 16) & 1u);   // round-to-nearest-even
    return (u16t)(u >> 16);
}

union V16U { v16bf v; v8bf h[2]; };

// Load a 16-element bf16 fragment: two contiguous 16B chunks at p and p+16
// (ISA 16-bit A/B layout: lane holds K = kg*8 + [0..7] and 16 + kg*8 + [0..7])
__device__ __forceinline__ v16bf ldfrag(const u16t* __restrict__ p) {
    V16U u;
    u.h[0] = *(const v8bf*)(p);
    u.h[1] = *(const v8bf*)(p + 16);
    return u.v;
}

// ---------------------------------------------------------------------------
// Generic bf16 WMMA GEMM:  C[M,N] = A[M,K] * W[N,K]^T + bias
// Block = 256 threads = 8 waves. Wave (wm,wn) computes rows [m0,m0+16) and
// cols [n0, n0+32) as two 16x16 tiles (A-fragment reused across both).
// Grid: (N/128, M/32). Requires M%32==0, N%128==0, K%32==0 (all shapes here).
// ---------------------------------------------------------------------------
template <bool OUT_BF16>
__global__ __launch_bounds__(256)
void gemm_wmma_bf16(const u16t* __restrict__ A, int lda,
                    const u16t* __restrict__ W, int ldw,
                    const float* __restrict__ bias,
                    void* __restrict__ Cout, long ldc, int K) {
    const int lane   = threadIdx.x & 31;
    const int waveId = threadIdx.x >> 5;
    const int wm = waveId >> 2;       // 0..1
    const int wn = waveId & 3;        // 0..3
    const int m0 = blockIdx.y * 32 + wm * 16;
    const int n0 = blockIdx.x * 128 + wn * 32;

    const int r  = lane & 15;         // row (A) / col (B) within tile
    const int kg = lane >> 4;         // K half-select

    const u16t* pa  = A + (long)(m0 + r) * lda + kg * 8;
    const u16t* pw0 = W + (long)(n0 + r) * ldw + kg * 8;
    const u16t* pw1 = W + (long)(n0 + 16 + r) * ldw + kg * 8;

    v8f acc0 = {0.f, 0.f, 0.f, 0.f, 0.f, 0.f, 0.f, 0.f};
    v8f acc1 = {0.f, 0.f, 0.f, 0.f, 0.f, 0.f, 0.f, 0.f};

#pragma unroll 4
    for (int k = 0; k < K; k += 32) {
        v16bf a  = ldfrag(pa + k);
        v16bf b0 = ldfrag(pw0 + k);
        v16bf b1 = ldfrag(pw1 + k);
        acc0 = __builtin_amdgcn_wmma_f32_16x16x32_bf16(
                   false, a, false, b0, (short)0, acc0, false, false);
        acc1 = __builtin_amdgcn_wmma_f32_16x16x32_bf16(
                   false, a, false, b1, (short)0, acc1, false, false);
    }

    // C/D layout: vgpr v, lane l -> m = v + 8*(l>>4), n = l&15
    const int nA = n0 + (lane & 15);
    const int nB = nA + 16;
    const int mb = m0 + (lane >> 4) * 8;
    const float bvA = bias ? bias[nA] : 0.f;
    const float bvB = bias ? bias[nB] : 0.f;

#pragma unroll
    for (int v = 0; v < 8; ++v) {
        float x0 = acc0[v] + bvA;
        float x1 = acc1[v] + bvB;
        long row = (long)(mb + v) * ldc;
        if (OUT_BF16) {
            ((u16t*)Cout)[row + nA] = f2bf(x0);
            ((u16t*)Cout)[row + nB] = f2bf(x1);
        } else {
            ((float*)Cout)[row + nA] = x0;
            ((float*)Cout)[row + nB] = x1;
        }
    }
}

// ---------------------------------------------------------------------------
// Elementwise / small kernels
// ---------------------------------------------------------------------------
__global__ void cvt_f32_to_bf16(const float* __restrict__ src,
                                u16t* __restrict__ dst, int n) {
    int i = blockIdx.x * blockDim.x + threadIdx.x;
    if (i < n) dst[i] = f2bf(src[i]);
}

// gather emb[captions] into bf16 (B,T,E)
__global__ void gather_embed_bf16(const int* __restrict__ captions,
                                  const float* __restrict__ emb,
                                  u16t* __restrict__ dst) {
    int i = blockIdx.x * blockDim.x + threadIdx.x;        // over B*T*E
    if (i >= Bc * Tc * Hc) return;
    int e  = i & (Hc - 1);
    int bt = i >> 9;                                      // b*T + t
    int tok = captions[bt];
    dst[i] = f2bf(emb[(long)tok * Hc + e]);
}

__global__ void init_h_zero(float* __restrict__ h, u16t* __restrict__ hbf) {
    int i = blockIdx.x * blockDim.x + threadIdx.x;        // over L*B*H
    if (i >= Lc * Bc * Hc) return;
    h[i] = 0.f;
    hbf[i] = 0;
}

// Fused attention: scores = tanh(q + keys_proj) . v_w (+v_b), softmax over S,
// context = sum_s w_s * features -> bf16; also copies x_t into rnn_in[:, :H].
__global__ __launch_bounds__(256)
void attn_kernel(const float* __restrict__ q,         // (B,H)
                 const float* __restrict__ keysp,     // (B,S,H)
                 const float* __restrict__ features,  // (B,S,H)
                 const float* __restrict__ v_w,       // (H)
                 const float* __restrict__ v_b,       // scalar
                 const u16t* __restrict__ x_bf,       // (B,T,H)
                 int t,
                 u16t* __restrict__ ctx_bf,           // (B,H)
                 u16t* __restrict__ rnn_bf) {         // (B,2H)
    const int b = blockIdx.x;
    const int tid = threadIdx.x;
    __shared__ float red[256];
    __shared__ float ws[Sc];

    const float* qb = q + b * Hc;
    const float* kp = keysp + (long)b * Sc * Hc;

    for (int s = 0; s < Sc; ++s) {
        float p = 0.f;
        for (int h = tid; h < Hc; h += 256)
            p += tanhf(qb[h] + kp[s * Hc + h]) * v_w[h];
        red[tid] = p;
        __syncthreads();
        for (int o = 128; o > 0; o >>= 1) {
            if (tid < o) red[tid] += red[tid + o];
            __syncthreads();
        }
        if (tid == 0) ws[s] = red[0] + v_b[0];
        __syncthreads();
    }
    if (tid == 0) {
        float mx = ws[0];
        for (int s = 1; s < Sc; ++s) mx = fmaxf(mx, ws[s]);
        float sum = 0.f;
        for (int s = 0; s < Sc; ++s) { float e = __expf(ws[s] - mx); ws[s] = e; sum += e; }
        float inv = 1.f / sum;
        for (int s = 0; s < Sc; ++s) ws[s] *= inv;
    }
    __syncthreads();

    const float* fb = features + (long)b * Sc * Hc;
    for (int k = tid; k < Hc; k += 256) {
        float acc = 0.f;
#pragma unroll 7
        for (int s = 0; s < Sc; ++s) acc += ws[s] * fb[s * Hc + k];
        ctx_bf[b * Hc + k] = f2bf(acc);
        rnn_bf[b * 2 * Hc + k] = x_bf[((long)b * Tc + t) * Hc + k];
    }
}

// torch GRUCell gate math; writes fp32 state in place + bf16 mirror.
__global__ void gru_gate(const float* __restrict__ gi,   // (B,3H)
                         const float* __restrict__ gh,   // (B,3H)
                         float* __restrict__ h,          // (B,H) in/out
                         u16t* __restrict__ hbf) {       // (B,H) out
    int i = blockIdx.x * blockDim.x + threadIdx.x;       // over B*H
    if (i >= Bc * Hc) return;
    int b = i >> 9, j = i & (Hc - 1);
    const float* gib = gi + b * 3 * Hc;
    const float* ghb = gh + b * 3 * Hc;
    float r = 1.f / (1.f + __expf(-(gib[j]          + ghb[j])));
    float z = 1.f / (1.f + __expf(-(gib[Hc + j]     + ghb[Hc + j])));
    float n = tanhf(gib[2 * Hc + j] + r * ghb[2 * Hc + j]);
    float hv = (1.f - z) * n + z * h[i];
    h[i] = hv;
    hbf[i] = f2bf(hv);
}

// ---------------------------------------------------------------------------
// Host orchestration
// ---------------------------------------------------------------------------
extern "C" void kernel_launch(void* const* d_in, const int* in_sizes, int n_in,
                              void* d_out, int out_size, void* d_ws, size_t ws_size,
                              hipStream_t stream) {
    const float* features = (const float*)d_in[0];
    const int*   captions = (const int*)d_in[1];
    const float* emb      = (const float*)d_in[2];
    const float* fc1_w    = (const float*)d_in[3];
    const float* fc1_b    = (const float*)d_in[4];
    const float* fc0_w    = (const float*)d_in[5];
    const float* fc0_b    = (const float*)d_in[6];
    const float* wq       = (const float*)d_in[7];
    const float* bq       = (const float*)d_in[8];
    const float* wk       = (const float*)d_in[9];
    const float* bk       = (const float*)d_in[10];
    const float* v_w      = (const float*)d_in[11];
    const float* v_b      = (const float*)d_in[12];
    const float* wih0     = (const float*)d_in[13];
    const float* whh0     = (const float*)d_in[14];
    const float* bih0     = (const float*)d_in[15];
    const float* bhh0     = (const float*)d_in[16];
    const float* wih_r    = (const float*)d_in[17];
    const float* whh_r    = (const float*)d_in[18];
    const float* bih_r    = (const float*)d_in[19];
    const float* bhh_r    = (const float*)d_in[20];
    const float* fc2_w    = (const float*)d_in[21];
    const float* fc2_b    = (const float*)d_in[22];
    float* out = (float*)d_out;

    // ---- workspace carve-up (256B aligned) ----
    char* wsp = (char*)d_ws;
    auto carve = [&](size_t bytes) {
        char* p = wsp;
        wsp += (bytes + 255) & ~(size_t)255;
        return p;
    };
    u16t* fc1_bf  = (u16t*)carve((size_t)Hc * Hc * 2);
    u16t* wq_bf   = (u16t*)carve((size_t)Hc * Hc * 2);
    u16t* wk_bf   = (u16t*)carve((size_t)Hc * Hc * 2);
    u16t* fc0_bf  = (u16t*)carve((size_t)Hc * Hc * 2);
    u16t* wih0_bf = (u16t*)carve((size_t)3 * Hc * 2 * Hc * 2);
    u16t* whh0_bf = (u16t*)carve((size_t)3 * Hc * Hc * 2);
    u16t* wihr_bf = (u16t*)carve((size_t)(Lc - 1) * 3 * Hc * Hc * 2);
    u16t* whhr_bf = (u16t*)carve((size_t)(Lc - 1) * 3 * Hc * Hc * 2);
    u16t* fc2_bf  = (u16t*)carve((size_t)Vc * Hc * 2);
    u16t* feat_bf = (u16t*)carve((size_t)Bc * Sc * Hc * 2);
    u16t* embx_bf = (u16t*)carve((size_t)Bc * Tc * Hc * 2);
    u16t* x_bf    = (u16t*)carve((size_t)Bc * Tc * Hc * 2);
    float* keysp  = (float*)carve((size_t)Bc * Sc * Hc * 4);
    float* h_f32  = (float*)carve((size_t)Lc * Bc * Hc * 4);
    u16t*  h_bf   = (u16t*)carve((size_t)Lc * Bc * Hc * 2);
    float* q_f32  = (float*)carve((size_t)Bc * Hc * 4);
    u16t*  ctx_bf = (u16t*)carve((size_t)Bc * Hc * 2);
    u16t*  rnn_bf = (u16t*)carve((size_t)Bc * 2 * Hc * 2);
    float* gi_f32 = (float*)carve((size_t)Bc * 3 * Hc * 4);
    float* gh_f32 = (float*)carve((size_t)Bc * 3 * Hc * 4);
    (void)ws_size; (void)n_in; (void)in_sizes; (void)out_size;

    auto cvt = [&](const float* s, u16t* d, int n) {
        cvt_f32_to_bf16<<<(n + 255) / 256, 256, 0, stream>>>(s, d, n);
    };
    auto gemm = [&](const u16t* A, int lda, const u16t* W, int ldw,
                    const float* bias, void* C, long ldc,
                    int M, int N, int K, bool outbf) {
        dim3 grid(N / 128, M / 32);
        if (outbf)
            gemm_wmma_bf16<true><<<grid, 256, 0, stream>>>(A, lda, W, ldw, bias, C, ldc, K);
        else
            gemm_wmma_bf16<false><<<grid, 256, 0, stream>>>(A, lda, W, ldw, bias, C, ldc, K);
    };

    // ---- preamble: bf16 weight/activation staging ----
    cvt(fc1_w, fc1_bf, Hc * Hc);
    cvt(wq, wq_bf, Hc * Hc);
    cvt(wk, wk_bf, Hc * Hc);
    cvt(fc0_w, fc0_bf, Hc * Hc);
    cvt(wih0, wih0_bf, 3 * Hc * 2 * Hc);
    cvt(whh0, whh0_bf, 3 * Hc * Hc);
    cvt(wih_r, wihr_bf, (Lc - 1) * 3 * Hc * Hc);
    cvt(whh_r, whhr_bf, (Lc - 1) * 3 * Hc * Hc);
    cvt(fc2_w, fc2_bf, Vc * Hc);
    cvt(features, feat_bf, Bc * Sc * Hc);
    gather_embed_bf16<<<(Bc * Tc * Hc + 255) / 256, 256, 0, stream>>>(captions, emb, embx_bf);
    init_h_zero<<<(Lc * Bc * Hc + 255) / 256, 256, 0, stream>>>(h_f32, h_bf);

    // x = emb[captions] @ fc1_w^T + fc1_b          (B*T, H) bf16
    gemm(embx_bf, Hc, fc1_bf, Hc, fc1_b, x_bf, Hc, Bc * Tc, Hc, Hc, true);
    // keys_proj = features @ wk^T + bk             (B*S, H) f32
    gemm(feat_bf, Hc, wk_bf, Hc, bk, keysp, Hc, Bc * Sc, Hc, Hc, false);

    const long ldout = (long)(Tc - 1) * Vc;         // row stride of (B, T-1, V)

    for (int t = 0; t < Tc - 1; ++t) {
        // q = h[L-1] @ wq^T + bq
        gemm(h_bf + (Lc - 1) * Bc * Hc, Hc, wq_bf, Hc, bq, q_f32, Hc,
             Bc, Hc, Hc, false);
        // attention + softmax + context; copy x_t into rnn_in[:, :H]
        attn_kernel<<<Bc, 256, 0, stream>>>(q_f32, keysp, features, v_w, v_b,
                                            x_bf, t, ctx_bf, rnn_bf);
        // ctx = context @ fc0_w^T + fc0_b -> rnn_in[:, H:2H] (bf16, ldc=2H)
        gemm(ctx_bf, Hc, fc0_bf, Hc, fc0_b, rnn_bf + Hc, 2 * Hc,
             Bc, Hc, Hc, true);
        // GRU layer 0
        gemm(rnn_bf, 2 * Hc, wih0_bf, 2 * Hc, bih0, gi_f32, 3 * Hc,
             Bc, 3 * Hc, 2 * Hc, false);
        gemm(h_bf, Hc, whh0_bf, Hc, bhh0, gh_f32, 3 * Hc,
             Bc, 3 * Hc, Hc, false);
        gru_gate<<<(Bc * Hc + 255) / 256, 256, 0, stream>>>(gi_f32, gh_f32,
                                                            h_f32, h_bf);
        // GRU layers 1..3
        for (int l = 1; l < Lc; ++l) {
            gemm(h_bf + (l - 1) * Bc * Hc, Hc,
                 wihr_bf + (size_t)(l - 1) * 3 * Hc * Hc, Hc,
                 bih_r + (size_t)(l - 1) * 3 * Hc, gi_f32, 3 * Hc,
                 Bc, 3 * Hc, Hc, false);
            gemm(h_bf + l * Bc * Hc, Hc,
                 whhr_bf + (size_t)(l - 1) * 3 * Hc * Hc, Hc,
                 bhh_r + (size_t)(l - 1) * 3 * Hc, gh_f32, 3 * Hc,
                 Bc, 3 * Hc, Hc, false);
            gru_gate<<<(Bc * Hc + 255) / 256, 256, 0, stream>>>(
                gi_f32, gh_f32, h_f32 + l * Bc * Hc, h_bf + l * Bc * Hc);
        }
        // out[:, t, :] = h[L-1] @ fc2_w^T + fc2_b  (f32, strided into d_out)
        gemm(h_bf + (Lc - 1) * Bc * Hc, Hc, fc2_bf, Hc, fc2_b,
             out + (long)t * Vc, ldout, Bc, Vc, Hc, false);
    }
}
